// GAT_11931419149196
// MI455X (gfx1250) — matrix operand order repaired and checked by reading, not compiled, not run on hardware
//
#include <hip/hip_runtime.h>
#include <hip/hip_bf16.h>
#include <math.h>

typedef __attribute__((ext_vector_type(16))) _Float16 v16h;
typedef __attribute__((ext_vector_type(8)))  float    v8f;

#define DEVI __device__ __forceinline__

constexpr int CB = 8;        // batch
constexpr int CN = 4096;     // nodes
constexpr int CL = 8;        // tokens per statement
constexpr int CE = 16384;    // edges per batch
constexpr int CH = 256;      // hidden
constexpr int CV = 32000;    // vocab
constexpr int COV = 512;     // out vocab
constexpr int CSTEPS = 6;
constexpr int CM = CB * CN;  // 32768 sequences / nodes total

// ---------------- workspace layout ----------------
constexpr size_t AL256(size_t x) { return (x + 255) & ~(size_t)255; }
constexpr size_t OFF_EMB  = 0;                                        // f16 [V,H]
constexpr size_t OFF_WI0  = OFF_EMB  + AL256((size_t)CV * CH * 2);    // f16 [4H,H] transposed
constexpr size_t OFF_WH0  = OFF_WI0  + AL256((size_t)4 * CH * CH * 2);
constexpr size_t OFF_WI1  = OFF_WH0  + AL256((size_t)4 * CH * CH * 2);
constexpr size_t OFF_WH1  = OFF_WI1  + AL256((size_t)4 * CH * CH * 2);
constexpr size_t OFF_WK   = OFF_WH1  + AL256((size_t)4 * CH * CH * 2); // f16 [H,H] transposed
constexpr size_t OFF_SE   = OFF_WK   + AL256((size_t)CH * CH * 2);     // f32 [M,H]
constexpr size_t OFF_KEYS = OFF_SE   + AL256((size_t)CM * CH * 4);     // f32 [M,H]
constexpr size_t OFF_SNEW = OFF_KEYS + AL256((size_t)CM * CH * 4);     // f32 [M,H]
constexpr size_t OFF_A1   = OFF_SNEW + AL256((size_t)CM * CH * 4);     // f32 [M]
constexpr size_t OFF_A2   = OFF_A1   + AL256((size_t)CM * 4);
constexpr size_t OFF_MB   = OFF_A2   + AL256((size_t)CM * 4);          // segment max
constexpr size_t OFF_SS   = OFF_MB   + AL256((size_t)CM * 4);          // segment sum
constexpr size_t OFF_EL   = OFF_SS   + AL256((size_t)CM * 4);          // f32 [B*E] logits/exp
constexpr size_t OFF_AT   = OFF_EL   + AL256((size_t)CB * CE * 4);     // f32 [6] edge-type scores

// ---------------- WMMA fragment helpers ----------------
// A fragment (16x32 f16, MxK), source row-major with leading dim `ld` halves.
// Lanes 0-15: row M=lane, K = k0+0..7 (v16h[0..7]) and k0+16..23 (v16h[8..15]).
// Lanes 16-31: row M=lane-16, K = k0+8..15 and k0+24..31.
DEVI v16h frag_a(const _Float16* p0, int ld) {
  int lane = threadIdx.x & 31;
  int lr = lane & 15, lg = lane >> 4;
  const _Float16* p = p0 + lr * ld + lg * 8;
  v16h a;
#pragma unroll
  for (int i = 0; i < 8; ++i) a[i] = p[i];
#pragma unroll
  for (int i = 0; i < 8; ++i) a[8 + i] = p[16 + i];
  return a;
}

// B fragment (32x16 f16, KxN) built from a TRANSPOSED weight matrix Wt[N_out][K]
// (row-major, leading dim `ld` halves). p0 points at Wt[j0][k0].
// Lanes 0-15: N=j0+lane, K = k0+0..7 and k0+16..23 ; lanes 16-31: N=j0+lane-16, K = k0+8..15 and k0+24..31.
DEVI v16h frag_b(const _Float16* p0, int ld) {
  int lane = threadIdx.x & 31;
  int lr = lane & 15, lg = lane >> 4;
  const _Float16* p = p0 + lr * ld + lg * 8;
  v16h b;
#pragma unroll
  for (int i = 0; i < 8; ++i) b[i] = p[i];
#pragma unroll
  for (int i = 0; i < 8; ++i) b[8 + i] = p[16 + i];
  return b;
}

#define WMMA_F16(A, Bm, C) \
  __builtin_amdgcn_wmma_f32_16x16x32_f16(false, (A), false, (Bm), (short)0, (C), false, false)

DEVI v8f vzero8() {
  v8f z;
#pragma unroll
  for (int i = 0; i < 8; ++i) z[i] = 0.f;
  return z;
}

DEVI float sigmf(float x) { return 1.f / (1.f + __expf(-x)); }

// sign-aware atomic float max (init with -inf)
DEVI void atomicMaxFloat(float* addr, float val) {
  if (val >= 0.f) atomicMax((int*)addr, __float_as_int(val));
  else            atomicMin((unsigned int*)addr, __float_as_uint(val));
}

// ---------------- prep kernels ----------------
__global__ void cvt_f16_kernel(const float* __restrict__ src, _Float16* __restrict__ dst, int n) {
  int i = blockIdx.x * 256 + threadIdx.x;
  if (i < n) dst[i] = (_Float16)src[i];
}

// W[rows][cols] (row-major f32) -> Wt[cols][rows] (row-major f16)
__global__ void transpose_f16_kernel(const float* __restrict__ W, _Float16* __restrict__ Wt,
                                     int rows, int cols) {
  int i = blockIdx.x * 256 + threadIdx.x;
  if (i < rows * cols) {
    int r = i / cols, c = i - r * cols;
    Wt[(size_t)c * rows + r] = (_Float16)W[i];
  }
}

// ---------------- fused 2-layer LSTM ----------------
// One workgroup (256 thr = 8 waves) owns 32 sequences. h-states ping-pong in
// LDS (f16) so the gate phase writes a different buffer than the matmul phase
// reads -> no barrier between matmul and apply. The jg/kk loops are kept
// rolled (#pragma unroll 1) so the scheduler cannot hoist all 64 B fragments
// at once and spill; per-iteration live set ~170 VGPRs.
DEVI void lstm_cell32(const _Float16* __restrict__ sx,      // input x (LDS f16)
                      const _Float16* __restrict__ sh_in,   // h_{t-1} (LDS f16)
                      _Float16* __restrict__ sh_out,        // h_t (LDS f16)
                      float* __restrict__ sc,               // c state (LDS f32)
                      const _Float16* __restrict__ wiT, const _Float16* __restrict__ whT,
                      const float* __restrict__ bias) {
  const int w = threadIdx.x >> 5;
  const int lane = threadIdx.x & 31;
  const int lr = lane & 15, lg = lane >> 4;

#pragma unroll 1
  for (int jg = 0; jg < 2; ++jg) {
    const int j0 = (w + jg * 8) * 16;
    v8f acc[2][4];
#pragma unroll
    for (int mt = 0; mt < 2; ++mt)
#pragma unroll
      for (int g = 0; g < 4; ++g) acc[mt][g] = vzero8();

#pragma unroll 1
    for (int kk = 0; kk < 8; ++kk) {
      const int k = kk * 32;
      v16h ax0 = frag_a(sx + k, CH);
      v16h ax1 = frag_a(sx + 16 * CH + k, CH);
      v16h ah0 = frag_a(sh_in + k, CH);
      v16h ah1 = frag_a(sh_in + 16 * CH + k, CH);
#pragma unroll
      for (int g = 0; g < 4; ++g) {
        v16h bi = frag_b(wiT + (size_t)(g * CH + j0) * CH + k, CH);
        v16h bh = frag_b(whT + (size_t)(g * CH + j0) * CH + k, CH);
        acc[0][g] = WMMA_F16(ax0, bi, acc[0][g]);
        acc[0][g] = WMMA_F16(ah0, bh, acc[0][g]);
        acc[1][g] = WMMA_F16(ax1, bi, acc[1][g]);
        acc[1][g] = WMMA_F16(ah1, bh, acc[1][g]);
      }
    }

    // gate phase: writes only sh_out / sc elements owned by this tile
    const int col = j0 + lr;
    const float bi_ = bias[col];
    const float bf_ = bias[CH + col];
    const float bg_ = bias[2 * CH + col];
    const float bo_ = bias[3 * CH + col];
#pragma unroll
    for (int mt = 0; mt < 2; ++mt) {
#pragma unroll
      for (int r = 0; r < 8; ++r) {
        const int row = mt * 16 + lg * 8 + r;
        const float zi = acc[mt][0][r] + bi_;
        const float zf = acc[mt][1][r] + bf_;
        const float zg = acc[mt][2][r] + bg_;
        const float zo = acc[mt][3][r] + bo_;
        const float cn = sigmf(zf) * sc[row * CH + col] + sigmf(zi) * tanhf(zg);
        sc[row * CH + col] = cn;
        sh_out[row * CH + col] = (_Float16)(sigmf(zo) * tanhf(cn));
      }
    }
  }
}

DEVI void stage_embed(_Float16* sh_e, const int* __restrict__ data,
                      const _Float16* __restrict__ emb, size_t seq0, int t) {
  const int tid = threadIdx.x;
  for (int i = tid; i < 1024; i += 256) {  // 32 rows x 32 uint4
    const int row = i >> 5, col = i & 31;
    const int tok = data[(seq0 + row) * CL + t];
    ((uint4*)sh_e)[row * 32 + col] = ((const uint4*)(emb + (size_t)tok * CH))[col];
  }
}

__global__ void __launch_bounds__(256)
lstm_kernel(const int* __restrict__ data, const _Float16* __restrict__ emb,
            const _Float16* __restrict__ wi0T, const _Float16* __restrict__ wh0T,
            const float* __restrict__ b0,
            const _Float16* __restrict__ wi1T, const _Float16* __restrict__ wh1T,
            const float* __restrict__ b1, float* __restrict__ se) {
  __shared__ __align__(16) _Float16 sh_e[32 * CH];
  __shared__ __align__(16) _Float16 h0A[32 * CH];
  __shared__ __align__(16) _Float16 h0B[32 * CH];
  __shared__ __align__(16) _Float16 h1A[32 * CH];
  __shared__ __align__(16) _Float16 h1B[32 * CH];
  __shared__ float sc0[32 * CH];
  __shared__ float sc1[32 * CH];
  const int tid = threadIdx.x;
  for (int i = tid; i < 32 * CH; i += 256) {
    h0A[i] = (_Float16)0.f;
    h1A[i] = (_Float16)0.f;
    sc0[i] = 0.f;
    sc1[i] = 0.f;
  }
  const size_t seq0 = (size_t)blockIdx.x * 32;

#pragma unroll 1
  for (int t = 0; t < CL; t += 2) {
    // step t: h*A -> h*B
    stage_embed(sh_e, data, emb, seq0, t);
    __syncthreads();
    lstm_cell32(sh_e, h0A, h0B, sc0, wi0T, wh0T, b0);
    __syncthreads();
    lstm_cell32(h0B, h1A, h1B, sc1, wi1T, wh1T, b1);
    __syncthreads();
    // step t+1: h*B -> h*A
    stage_embed(sh_e, data, emb, seq0, t + 1);
    __syncthreads();
    lstm_cell32(sh_e, h0B, h0A, sc0, wi0T, wh0T, b0);
    __syncthreads();
    lstm_cell32(h0A, h1B, h1A, sc1, wi1T, wh1T, b1);
    __syncthreads();
  }
  // after 8 steps (even), last layer-1 output is in h1A
  for (int i = tid; i < 32 * CH; i += 256) se[seq0 * CH + i] = (float)h1A[i];
}

// ---------------- GAT: keys = se @ Wk + bk ----------------
__global__ void __launch_bounds__(256)
keys_gemm_kernel(const float* __restrict__ se, const _Float16* __restrict__ wkT,
                 const float* __restrict__ bk, float* __restrict__ keys) {
  __shared__ __align__(16) _Float16 sa[32 * CH];
  const int tid = threadIdx.x;
  const size_t base = (size_t)blockIdx.x * 32 * CH;
  for (int i = tid; i < 32 * CH; i += 256) sa[i] = (_Float16)se[base + i];
  __syncthreads();

  const int w = tid >> 5, lane = tid & 31;
  v8f acc[2][2];
#pragma unroll
  for (int jg = 0; jg < 2; ++jg)
#pragma unroll
    for (int mt = 0; mt < 2; ++mt) acc[jg][mt] = vzero8();

#pragma unroll 1
  for (int jg = 0; jg < 2; ++jg) {
    const int j0 = (w + jg * 8) * 16;
#pragma unroll 1
    for (int kk = 0; kk < 8; ++kk) {
      const int k = kk * 32;
      v16h a0 = frag_a(sa + k, CH);
      v16h a1 = frag_a(sa + 16 * CH + k, CH);
      v16h b = frag_b(wkT + (size_t)j0 * CH + k, CH);
      acc[jg][0] = WMMA_F16(a0, b, acc[jg][0]);
      acc[jg][1] = WMMA_F16(a1, b, acc[jg][1]);
    }
  }
  const int lr = lane & 15, lg = lane >> 4;
#pragma unroll
  for (int jg = 0; jg < 2; ++jg) {
    const int col = (w + jg * 8) * 16 + lr;
    const float bb = bk[col];
#pragma unroll
    for (int mt = 0; mt < 2; ++mt)
#pragma unroll
      for (int r = 0; r < 8; ++r) {
        const int row = mt * 16 + lg * 8 + r;
        keys[base + (size_t)row * CH + col] = acc[jg][mt][r] + bb;
      }
  }
}

// ---------------- GAT scalar pieces ----------------
__global__ void at_kernel(const float* __restrict__ edge_table, const float* __restrict__ Wa,
                          float* __restrict__ atv) {
  const int w = threadIdx.x >> 5, lane = threadIdx.x & 31;
  if (w < 6) {
    float s = 0.f;
    for (int i = lane; i < CH; i += 32) s += edge_table[w * CH + i] * Wa[i];
    for (int o = 16; o; o >>= 1) s += __shfl_xor(s, o, 32);
    if (lane == 0) atv[w] = s;
  }
}

__global__ void node_score_kernel(const float* __restrict__ keys, const float* __restrict__ Wa,
                                  float* __restrict__ a1, float* __restrict__ a2) {
  const int w = threadIdx.x >> 5, lane = threadIdx.x & 31;
  const size_t node = (size_t)blockIdx.x * 8 + w;
  const float* kp = keys + node * CH;
  float s1 = 0.f, s2 = 0.f;
  for (int i = lane; i < CH; i += 32) {
    const float k = kp[i];
    s1 += k * Wa[i];
    s2 += k * Wa[CH + i];
  }
  for (int o = 16; o; o >>= 1) {
    s1 += __shfl_xor(s1, o, 32);
    s2 += __shfl_xor(s2, o, 32);
  }
  if (lane == 0) { a1[node] = s1; a2[node] = s2; }
}

__global__ void init_step_kernel(float* __restrict__ senew, float* __restrict__ m,
                                 float* __restrict__ ssum) {
  const size_t i = (size_t)blockIdx.x * 256 + threadIdx.x;
  if (i < (size_t)CM * CH) senew[i] = 0.f;
  if (i < (size_t)CM) { m[i] = -__builtin_inff(); ssum[i] = 0.f; }
}

__global__ void edge_logit_kernel(const int* __restrict__ src, const int* __restrict__ dst,
                                  const int* __restrict__ et, const float* __restrict__ a1,
                                  const float* __restrict__ a2, const float* __restrict__ atv,
                                  const float* __restrict__ ba, float* __restrict__ elog,
                                  float* __restrict__ m) {
  const int e = blockIdx.x * 256 + threadIdx.x;  // < B*E
  const int b = e >> 14;                         // E = 16384
  const int s = src[e], d = dst[e], t = et[e];
  float x = a1[b * CN + s] + atv[t] + a2[b * CN + d] + ba[0];
  x = fmaxf(0.2f * x, x);  // leaky relu
  elog[e] = x;
  atomicMaxFloat(&m[b * CN + d], x);
}

__global__ void edge_exp_kernel(const int* __restrict__ dst, const float* __restrict__ m,
                                float* __restrict__ elog, float* __restrict__ ssum) {
  const int e = blockIdx.x * 256 + threadIdx.x;
  const int b = e >> 14;
  const int d = dst[e];
  const float ex = __expf(elog[e] - m[b * CN + d]);
  elog[e] = ex;
  atomicAdd(&ssum[b * CN + d], ex);
}

__global__ void edge_agg_kernel(const int* __restrict__ src, const int* __restrict__ dst,
                                const int* __restrict__ et, const float* __restrict__ elog,
                                const float* __restrict__ ssum, const float* __restrict__ keys,
                                const float* __restrict__ edge_table, float* __restrict__ senew) {
  const int w = threadIdx.x >> 5, lane = threadIdx.x & 31;
  const int e = blockIdx.x * 8 + w;  // one wave per edge
  const int b = e >> 14;
  const int s = src[e], d = dst[e], t = et[e];
  const float wgt = elog[e] / ssum[b * CN + d];
  const float* kp = keys + ((size_t)b * CN + s) * CH;
  const float* ep = edge_table + (size_t)t * CH;
  float* op = senew + ((size_t)b * CN + d) * CH;
  for (int i = lane; i < CH; i += 32) atomicAdd(&op[i], (kp[i] + ep[i]) * wgt);
}

__global__ void se_update_kernel(float* __restrict__ se, const float* __restrict__ senew,
                                 const int* __restrict__ steps, int step) {
  const size_t i = (size_t)blockIdx.x * 256 + threadIdx.x;  // < M*H
  const int b = (int)(i >> 20);                             // N*H = 2^20
  if (step < steps[b]) se[i] = senew[i];
}

// ---------------- output head ----------------
__global__ void output_kernel(const float* __restrict__ se, const int* __restrict__ exit_index,
                              const float* __restrict__ Wo, const float* __restrict__ bo,
                              float* __restrict__ out) {
  const int b = blockIdx.x, tid = threadIdx.x;
  const int node = exit_index[b];
  const float* fp = se + ((size_t)b * CN + node) * CH;
  float acc0 = 0.f, acc1 = 0.f;
  for (int k = 0; k < CH; ++k) {
    const float f = fp[k];
    acc0 += f * Wo[(size_t)k * COV + tid];
    acc1 += f * Wo[(size_t)k * COV + tid + 256];
  }
  out[(size_t)b * COV + tid] = acc0 + bo[tid];
  out[(size_t)b * COV + tid + 256] = acc1 + bo[tid + 256];
}

// ---------------- host launcher ----------------
extern "C" void kernel_launch(void* const* d_in, const int* in_sizes, int n_in,
                              void* d_out, int out_size, void* d_ws, size_t ws_size,
                              hipStream_t stream) {
  const int*   data    = (const int*)d_in[0];
  const int*   srcI    = (const int*)d_in[1];
  const int*   dstI    = (const int*)d_in[2];
  const int*   etI     = (const int*)d_in[3];
  const int*   steps   = (const int*)d_in[4];
  const int*   exitI   = (const int*)d_in[5];
  const float* embed   = (const float*)d_in[6];
  const float* Wi0     = (const float*)d_in[7];
  const float* Wh0     = (const float*)d_in[8];
  const float* b0      = (const float*)d_in[9];
  const float* Wi1     = (const float*)d_in[10];
  const float* Wh1     = (const float*)d_in[11];
  const float* b1      = (const float*)d_in[12];
  const float* Wk      = (const float*)d_in[13];
  const float* bk      = (const float*)d_in[14];
  const float* Wa      = (const float*)d_in[15];
  const float* ba      = (const float*)d_in[16];
  const float* etab    = (const float*)d_in[17];
  const float* Wo      = (const float*)d_in[18];
  const float* bo      = (const float*)d_in[19];
  float* out = (float*)d_out;

  char* ws = (char*)d_ws;
  _Float16* emb16 = (_Float16*)(ws + OFF_EMB);
  _Float16* wi0T  = (_Float16*)(ws + OFF_WI0);
  _Float16* wh0T  = (_Float16*)(ws + OFF_WH0);
  _Float16* wi1T  = (_Float16*)(ws + OFF_WI1);
  _Float16* wh1T  = (_Float16*)(ws + OFF_WH1);
  _Float16* wkT   = (_Float16*)(ws + OFF_WK);
  float* se    = (float*)(ws + OFF_SE);
  float* keys  = (float*)(ws + OFF_KEYS);
  float* senew = (float*)(ws + OFF_SNEW);
  float* a1    = (float*)(ws + OFF_A1);
  float* a2    = (float*)(ws + OFF_A2);
  float* mbuf  = (float*)(ws + OFF_MB);
  float* ssum  = (float*)(ws + OFF_SS);
  float* elog  = (float*)(ws + OFF_EL);
  float* atv   = (float*)(ws + OFF_AT);

  // 1) weight / embedding prep
  cvt_f16_kernel<<<(CV * CH + 255) / 256, 256, 0, stream>>>(embed, emb16, CV * CH);
  transpose_f16_kernel<<<(CH * 4 * CH + 255) / 256, 256, 0, stream>>>(Wi0, wi0T, CH, 4 * CH);
  transpose_f16_kernel<<<(CH * 4 * CH + 255) / 256, 256, 0, stream>>>(Wh0, wh0T, CH, 4 * CH);
  transpose_f16_kernel<<<(CH * 4 * CH + 255) / 256, 256, 0, stream>>>(Wi1, wi1T, CH, 4 * CH);
  transpose_f16_kernel<<<(CH * 4 * CH + 255) / 256, 256, 0, stream>>>(Wh1, wh1T, CH, 4 * CH);
  transpose_f16_kernel<<<(CH * CH + 255) / 256, 256, 0, stream>>>(Wk, wkT, CH, CH);

  // 2) fused embedding + 2-layer LSTM -> se [M,H]
  lstm_kernel<<<CM / 32, 256, 0, stream>>>(data, emb16, wi0T, wh0T, b0, wi1T, wh1T, b1, se);

  // 3) constant edge-type scores
  at_kernel<<<1, 256, 0, stream>>>(etab, Wa, atv);

  // 4) GAT message-passing steps
  for (int step = 0; step < CSTEPS; ++step) {
    init_step_kernel<<<(CM * CH) / 256, 256, 0, stream>>>(senew, mbuf, ssum);
    keys_gemm_kernel<<<CM / 32, 256, 0, stream>>>(se, wkT, bk, keys);
    node_score_kernel<<<CM / 8, 256, 0, stream>>>(keys, Wa, a1, a2);
    edge_logit_kernel<<<(CB * CE) / 256, 256, 0, stream>>>(srcI, dstI, etI, a1, a2, atv, ba,
                                                           elog, mbuf);
    edge_exp_kernel<<<(CB * CE) / 256, 256, 0, stream>>>(dstI, mbuf, elog, ssum);
    edge_agg_kernel<<<(CB * CE) / 8, 256, 0, stream>>>(srcI, dstI, etI, elog, ssum, keys, etab,
                                                       senew);
    se_update_kernel<<<(CM * CH) / 256, 256, 0, stream>>>(se, senew, steps, step);
  }

  // 5) output head
  output_kernel<<<CB, 256, 0, stream>>>(se, exitI, Wo, bo, out);
}
